// DualAttentionModule_62371515073031
// MI455X (gfx1250) — matrix operand order repaired and checked by reading, not compiled, not run on hardware
//
#include <hip/hip_runtime.h>
#include <hip/hip_bf16.h>

typedef __attribute__((ext_vector_type(16))) _Float16 v16h;
typedef __attribute__((ext_vector_type(8)))  _Float16 v8h;
typedef __attribute__((ext_vector_type(8)))  float    v8f;

#define WMMA_F16(a, b, c) \
  __builtin_amdgcn_wmma_f32_16x16x32_f16(false, (a), false, (b), (short)0, (c), false, false)

// ---------------------------------------------------------------------------
// Fragment loaders per CDNA5 ISA layouts (CDNA5 ISA, 16-bit A 16x32 / C,D 16x16)
// A (16x32, MxK): lane m=lane&15; element e -> K = ((e>=8)?16:0) + hi*8 + (e&7)
// B (32x16, KxN): lane col n=lane&15; element e -> K = ((e>=8)?16:0) + hi*8 + (e&7)
// C/D (16x16 f32): element r -> (M = hi*8 + r, N = lane&15)
// ---------------------------------------------------------------------------
__device__ __forceinline__ v16h load_A16x32(const _Float16* __restrict__ A, int lda,
                                            int kbase, int lane) {
  const int hi = (lane >> 4) & 1;
  const _Float16* p = A + (size_t)(lane & 15) * lda + kbase + hi * 8;
  v8h lo = *(const v8h*)(p);
  v8h hh = *(const v8h*)(p + 16);
  v16h a;
#pragma unroll
  for (int j = 0; j < 8; ++j) { a[j] = lo[j]; a[j + 8] = hh[j]; }
  return a;
}

// Bt is the B operand stored transposed: Bt[col][k] contiguous in k.
__device__ __forceinline__ v16h load_B32x16(const _Float16* __restrict__ Bt, int ldb,
                                            int kbase, int lane) {
  const int hi = (lane >> 4) & 1;
  const _Float16* p = Bt + (size_t)(lane & 15) * ldb + kbase + hi * 8;
  v8h x0 = *(const v8h*)(p);
  v8h x1 = *(const v8h*)(p + 16);
  v16h b;
#pragma unroll
  for (int j = 0; j < 8; ++j) { b[j] = x0[j]; b[j + 8] = x1[j]; }
  return b;
}

// ---------------------------------------------------------------------------
// Generic WMMA GEMM: Out = A(f16,[M,K]) * B + optional bias.
// B supplied transposed: Bt[N][K]. 256 threads = 8 waves.
// Each wave computes 16(M) x 32(N) (2 accumulators sharing one A fragment);
// block tile = 32(M) x 128(N).
// ---------------------------------------------------------------------------
template <bool OUT_F16, bool TRANS_OUT, bool HAS_BIAS>
__global__ __launch_bounds__(256) void gemm_wmma_f16(
    const _Float16* __restrict__ A, size_t strideA,
    const _Float16* __restrict__ Bt, size_t strideB,
    const float* __restrict__ bias,
    void* __restrict__ Out, size_t strideO,
    int M, int N, int K) {
  const int lane = threadIdx.x & 31;
  const int w = threadIdx.x >> 5;       // 0..7
  const int n32 = w & 3;                // 4 N-subtiles of 32
  const int m16 = w >> 2;               // 2 M-subtiles of 16
  const int z = blockIdx.z;
  const int mBase = blockIdx.y * 32 + m16 * 16;
  const int nBase = blockIdx.x * 128 + n32 * 32;

  const _Float16* Ab = A + (size_t)z * strideA + (size_t)mBase * K;
  const _Float16* B0 = Bt + (size_t)z * strideB + (size_t)nBase * K;
  const _Float16* B1 = B0 + (size_t)16 * K;

  const int l15 = lane & 15;
  v8f acc0 = {}, acc1 = {};
  for (int k = 0; k < K; k += 32) {
    v16h a  = load_A16x32(Ab, K, k, lane);
    v16h b0 = load_B32x16(B0, K, k, lane);
    v16h b1 = load_B32x16(B1, K, k, lane);
    acc0 = WMMA_F16(a, b0, acc0);
    acc1 = WMMA_F16(a, b1, acc1);
    if (k + 256 < K) {  // prefetch 8 k-steps (256 elements) ahead into GL2
      __builtin_prefetch(Ab + (size_t)l15 * K + k + 256, 0, 0);
      __builtin_prefetch(B0 + (size_t)l15 * K + k + 256, 0, 0);
      __builtin_prefetch(B1 + (size_t)l15 * K + k + 256, 0, 0);
    }
  }

  const int hi = (lane >> 4) & 1;
  if (HAS_BIAS) {
#pragma unroll
    for (int r = 0; r < 8; ++r) {
      float bb = bias[mBase + hi * 8 + r];
      acc0[r] += bb;
      acc1[r] += bb;
    }
  }
  if (OUT_F16) {
    _Float16* O = (_Float16*)Out + (size_t)z * strideO;
    if (TRANS_OUT) {
      _Float16* p0 = O + (size_t)(nBase + l15) * M + mBase + hi * 8;
      _Float16* p1 = O + (size_t)(nBase + 16 + l15) * M + mBase + hi * 8;
#pragma unroll
      for (int r = 0; r < 8; ++r) { p0[r] = (_Float16)acc0[r]; p1[r] = (_Float16)acc1[r]; }
    } else {
#pragma unroll
      for (int r = 0; r < 8; ++r) {
        O[(size_t)(mBase + hi * 8 + r) * N + nBase + l15] = (_Float16)acc0[r];
        O[(size_t)(mBase + hi * 8 + r) * N + nBase + 16 + l15] = (_Float16)acc1[r];
      }
    }
  } else {
    float* O = (float*)Out + (size_t)z * strideO;
#pragma unroll
    for (int r = 0; r < 8; ++r) {
      O[(size_t)(mBase + hi * 8 + r) * N + nBase + l15] = acc0[r];
      O[(size_t)(mBase + hi * 8 + r) * N + nBase + 16 + l15] = acc1[r];
    }
  }
}

// ---------------------------------------------------------------------------
// PAM flash attention: per block, 16 query rows; stream m in 128-wide tiles.
// All 8 waves compute one 16x16 S subtile each (S = q^T k, K=64) into LDS;
// softmax is parallelized across all 256 threads (16 rows x 16 segments);
// each wave then applies P @ v^T for its 64 output channels.
// The [4096 x 4096] energy matrix never touches HBM.
// ---------------------------------------------------------------------------
__global__ __launch_bounds__(256) void pam_flash(
    const _Float16* __restrict__ qT,   // [B][N][Cq]
    const _Float16* __restrict__ kT,   // [B][N][Cq]
    const _Float16* __restrict__ vH,   // [B][C][N]
    _Float16* __restrict__ pamH,       // [B][N][C]
    int C, int N, int Cq) {
  __shared__ float S[16][128];
  __shared__ _Float16 P[16][128];
  __shared__ float Smax[16][16];
  __shared__ float Ssum[16][16];
  __shared__ float rmax[16], rsum[16], ralpha[16], rnm[16];

  const int lane = threadIdx.x & 31;
  const int w = threadIdx.x >> 5;
  const int hi = (lane >> 4) & 1;
  const int l15 = lane & 15;
  const int b = blockIdx.y;
  const int nBase = blockIdx.x * 16;
  const int cw = w * 64;               // channel range owned by this wave
  const int tr = threadIdx.x & 15;     // softmax: row
  const int ts = threadIdx.x >> 4;     // softmax: 8-col segment 0..15

  const _Float16* qb = qT + ((size_t)b * N + nBase) * Cq;
  const _Float16* kb = kT + (size_t)b * N * Cq;
  const _Float16* vb = vH + (size_t)b * C * N;

  // Q fragments are reused for the whole m-loop.
  v16h aq0 = load_A16x32(qb, Cq, 0, lane);
  v16h aq1 = load_A16x32(qb, Cq, 32, lane);

  v8f zero = {};
  v8f acc[4] = {zero, zero, zero, zero};

  if (threadIdx.x < 16) { rmax[threadIdx.x] = -1e30f; rsum[threadIdx.x] = 0.f; }
  __syncthreads();

  const _Float16* Pp = &P[0][0];

  for (int mt = 0; mt < N; mt += 128) {
    {  // S subtile for this wave: columns [w*16, w*16+16)
      const _Float16* kt = kb + (size_t)(mt + w * 16) * Cq;
      v16h b0 = load_B32x16(kt, Cq, 0, lane);
      v16h b1 = load_B32x16(kt, Cq, 32, lane);
      v8f s = {};
      s = WMMA_F16(aq0, b0, s);
      s = WMMA_F16(aq1, b1, s);
#pragma unroll
      for (int r = 0; r < 8; ++r) S[hi * 8 + r][w * 16 + l15] = s[r];
      if (mt + 128 < N)
        __builtin_prefetch(kb + (size_t)(mt + 128 + w * 16) * Cq, 0, 0);
    }
    __syncthreads();

    {  // per-segment maxima (all 256 threads)
      const float* sr = &S[tr][ts * 8];
      float m = sr[0];
#pragma unroll
      for (int j = 1; j < 8; ++j) m = fmaxf(m, sr[j]);
      Smax[tr][ts] = m;
    }
    __syncthreads();

    if (threadIdx.x < 16) {  // combine: new running max + rescale factor
      const int i = threadIdx.x;
      float tm = Smax[i][0];
#pragma unroll
      for (int j = 1; j < 16; ++j) tm = fmaxf(tm, Smax[i][j]);
      float nm = fmaxf(rmax[i], tm);
      ralpha[i] = __expf(rmax[i] - nm);
      rnm[i] = nm;
      rmax[i] = nm;
    }
    __syncthreads();

    {  // exp + partial sums (all 256 threads) + accumulator rescale
      const float nm = rnm[tr];
      const float* sr = &S[tr][ts * 8];
      _Float16* pr = &P[tr][ts * 8];
      float ps = 0.f;
#pragma unroll
      for (int j = 0; j < 8; ++j) {
        float p = __expf(sr[j] - nm);
        pr[j] = (_Float16)p;
        ps += p;
      }
      Ssum[tr][ts] = ps;
#pragma unroll
      for (int r = 0; r < 8; ++r) {
        float a_ = ralpha[hi * 8 + r];
        acc[0][r] *= a_; acc[1][r] *= a_; acc[2][r] *= a_; acc[3][r] *= a_;
      }
    }
    __syncthreads();

    if (threadIdx.x < 16) {  // running denominator (read after final barrier)
      const int i = threadIdx.x;
      float ps = Ssum[i][0];
#pragma unroll
      for (int j = 1; j < 16; ++j) ps += Ssum[i][j];
      rsum[i] = rsum[i] * ralpha[i] + ps;
    }

    // apply: acc += P(16x128) @ v^T(128 x 64 channels of this wave)
    v16h ap0 = load_A16x32(Pp, 128, 0, lane);
    v16h ap1 = load_A16x32(Pp, 128, 32, lane);
    v16h ap2 = load_A16x32(Pp, 128, 64, lane);
    v16h ap3 = load_A16x32(Pp, 128, 96, lane);
#pragma unroll
    for (int t = 0; t < 4; ++t) {
      const _Float16* vt = vb + (size_t)(cw + t * 16) * N + mt;  // rows=channels, ld=N
      v16h b0 = load_B32x16(vt, N, 0, lane);
      v16h b1 = load_B32x16(vt, N, 32, lane);
      v16h b2 = load_B32x16(vt, N, 64, lane);
      v16h b3 = load_B32x16(vt, N, 96, lane);
      acc[t] = WMMA_F16(ap0, b0, acc[t]);
      acc[t] = WMMA_F16(ap1, b1, acc[t]);
      acc[t] = WMMA_F16(ap2, b2, acc[t]);
      acc[t] = WMMA_F16(ap3, b3, acc[t]);
      if (mt + 128 < N) __builtin_prefetch(vt + 128, 0, 0);
    }
    __syncthreads();  // protect S/P (and order rsum) for next iteration / exit
  }

#pragma unroll
  for (int t = 0; t < 4; ++t) {
#pragma unroll
    for (int r = 0; r < 8; ++r) {
      float inv = 1.f / rsum[hi * 8 + r];
      int n = nBase + hi * 8 + r;
      pamH[((size_t)b * N + n) * C + cw + t * 16 + l15] = (_Float16)(acc[t][r] * inv);
    }
  }
}

// ---------------------------------------------------------------------------
// Small helper kernels
// ---------------------------------------------------------------------------
__global__ void f32_to_f16(const float* __restrict__ in, _Float16* __restrict__ out, size_t n) {
  size_t i = (size_t)blockIdx.x * blockDim.x + threadIdx.x;
  if (i < n) out[i] = (_Float16)in[i];
}

// in: f32 [B][C][N] -> out: f16 [B][N][C]
__global__ __launch_bounds__(256) void transpose_f32_to_f16(
    const float* __restrict__ in, _Float16* __restrict__ out, int C, int N) {
  __shared__ float tile[32][33];
  const int b = blockIdx.z;
  const int n0 = blockIdx.x * 32, c0 = blockIdx.y * 32;
  const float* src = in + (size_t)b * C * N;
  _Float16* dst = out + (size_t)b * N * C;
  for (int i = threadIdx.y; i < 32; i += 8)
    tile[i][threadIdx.x] = src[(size_t)(c0 + i) * N + n0 + threadIdx.x];
  __syncthreads();
  for (int i = threadIdx.y; i < 32; i += 8)
    dst[(size_t)(n0 + i) * C + c0 + threadIdx.x] = (_Float16)tile[threadIdx.x][i];
}

// softmax over d of (rowmax - e) == exp(rmin - e)/sum : only rmin needed.
__global__ __launch_bounds__(256) void cam_softmax(const float* __restrict__ energy,
                                                   _Float16* __restrict__ attn, int C) {
  const int b = blockIdx.y, c = blockIdx.x, tid = threadIdx.x;
  const float* e = energy + ((size_t)b * C + c) * C;
  _Float16* a = attn + ((size_t)b * C + c) * C;
  __shared__ float red[256];
  float m = 3.4e38f;
  for (int d = tid; d < C; d += 256) m = fminf(m, e[d]);
  red[tid] = m; __syncthreads();
  for (int s = 128; s > 0; s >>= 1) {
    if (tid < s) red[tid] = fminf(red[tid], red[tid + s]);
    __syncthreads();
  }
  const float rmin = red[0];
  __syncthreads();
  float sum = 0.f;
  for (int d = tid; d < C; d += 256) sum += __expf(rmin - e[d]);
  red[tid] = sum; __syncthreads();
  for (int s = 128; s > 0; s >>= 1) {
    if (tid < s) red[tid] += red[tid + s];
    __syncthreads();
  }
  const float inv = 1.f / red[0];
  for (int d = tid; d < C; d += 256) a[d] = (_Float16)(__expf(rmin - e[d]) * inv);
}

// featT[b][n][c] = gamma_p*pam[b][n][c] + gamma_c*cam[b][c][n] + 2*x[b][c][n]
__global__ __launch_bounds__(256) void combine_kernel(
    const _Float16* __restrict__ pamH, const _Float16* __restrict__ camH,
    const float* __restrict__ x, const float* __restrict__ gp, const float* __restrict__ gc,
    _Float16* __restrict__ featT, int C, int N) {
  __shared__ float tile[32][33];
  const int b = blockIdx.z;
  const int n0 = blockIdx.x * 32, c0 = blockIdx.y * 32;
  const float g_p = gp[0], g_c = gc[0];
  const _Float16* cam = camH + (size_t)b * C * N;
  const float* xs = x + (size_t)b * C * N;
  const _Float16* pam = pamH + (size_t)b * N * C;
  _Float16* dst = featT + (size_t)b * N * C;
  for (int i = threadIdx.y; i < 32; i += 8) {
    size_t idx = (size_t)(c0 + i) * N + n0 + threadIdx.x;
    tile[i][threadIdx.x] = g_c * (float)cam[idx] + 2.0f * xs[idx];
  }
  __syncthreads();
  for (int i = threadIdx.y; i < 32; i += 8) {
    size_t o = (size_t)(n0 + i) * C + c0 + threadIdx.x;
    dst[o] = (_Float16)(tile[threadIdx.x][i] + g_p * (float)pam[o]);
  }
}

__global__ __launch_bounds__(256) void bn_stats(const float* __restrict__ convO,
                                                float* __restrict__ mean,
                                                float* __restrict__ rstd,
                                                int B, int C, int N) {
  const int o = blockIdx.x, tid = threadIdx.x;
  float s = 0.f, s2 = 0.f;
  for (int b = 0; b < B; ++b) {
    const float* p = convO + ((size_t)b * C + o) * N;
    for (int n = tid; n < N; n += 256) { float v = p[n]; s += v; s2 += v * v; }
  }
  __shared__ float rs[256], rq[256];
  rs[tid] = s; rq[tid] = s2; __syncthreads();
  for (int st = 128; st > 0; st >>= 1) {
    if (tid < st) { rs[tid] += rs[tid + st]; rq[tid] += rq[tid + st]; }
    __syncthreads();
  }
  if (tid == 0) {
    float inv = 1.f / (float)(B * N);
    float mu = rs[0] * inv;
    float var = rq[0] * inv - mu * mu;
    mean[o] = mu;
    rstd[o] = rsqrtf(var + 1e-5f);
  }
}

__global__ void bn_apply(const float* __restrict__ convO,
                         const float* __restrict__ mean, const float* __restrict__ rstd,
                         const float* __restrict__ g, const float* __restrict__ bta,
                         float* __restrict__ out, int C, int N, size_t total) {
  size_t i = (size_t)blockIdx.x * blockDim.x + threadIdx.x;
  if (i >= total) return;
  int o = (int)((i / (size_t)N) % (size_t)C);
  float v = (convO[i] - mean[o]) * rstd[o] * g[o] + bta[o];
  out[i] = v > 0.f ? v : 0.f;
}

// ---------------------------------------------------------------------------
extern "C" void kernel_launch(void* const* d_in, const int* in_sizes, int n_in,
                              void* d_out, int out_size, void* d_ws, size_t ws_size,
                              hipStream_t stream) {
  (void)in_sizes; (void)n_in; (void)out_size; (void)ws_size;
  const int B = 4, C = 512, N = 4096, Cq = 64;

  const float* x    = (const float*)d_in[0];
  const float* Wq   = (const float*)d_in[1];
  const float* bq   = (const float*)d_in[2];
  const float* Wk   = (const float*)d_in[3];
  const float* bk   = (const float*)d_in[4];
  const float* Wv   = (const float*)d_in[5];
  const float* bv   = (const float*)d_in[6];
  const float* gp   = (const float*)d_in[7];
  const float* gc   = (const float*)d_in[8];
  const float* Wout = (const float*)d_in[9];
  const float* bng  = (const float*)d_in[10];
  const float* bnb  = (const float*)d_in[11];

  char* base = (char*)d_ws;
  size_t off = 0;
  auto alloc = [&](size_t bytes) {
    void* p = base + off;
    off = (off + bytes + 255) & ~(size_t)255;
    return p;
  };

  _Float16* xh    = (_Float16*)alloc((size_t)B * C * N * 2);
  _Float16* xhT   = (_Float16*)alloc((size_t)B * N * C * 2);
  _Float16* WqH   = (_Float16*)alloc((size_t)Cq * C * 2);
  _Float16* WkH   = (_Float16*)alloc((size_t)Cq * C * 2);
  _Float16* WvH   = (_Float16*)alloc((size_t)C * C * 2);
  _Float16* WoH   = (_Float16*)alloc((size_t)C * C * 2);
  _Float16* qT    = (_Float16*)alloc((size_t)B * N * Cq * 2);
  _Float16* kTm   = (_Float16*)alloc((size_t)B * N * Cq * 2);
  _Float16* vH    = (_Float16*)alloc((size_t)B * C * N * 2);
  _Float16* pamH  = (_Float16*)alloc((size_t)B * N * C * 2);
  float*    ener  = (float*)  alloc((size_t)B * C * C * 4);
  _Float16* attnH = (_Float16*)alloc((size_t)B * C * C * 2);
  _Float16* camH  = (_Float16*)alloc((size_t)B * C * N * 2);
  _Float16* featT = (_Float16*)alloc((size_t)B * N * C * 2);
  float*    convO = (float*)  alloc((size_t)B * C * N * 4);
  float*    mean  = (float*)  alloc((size_t)C * 4);
  float*    rstd  = (float*)  alloc((size_t)C * 4);

  // --- precision converts / transposes ---
  size_t nx = (size_t)B * C * N;
  f32_to_f16<<<dim3((unsigned)((nx + 255) / 256)), 256, 0, stream>>>(x, xh, nx);
  f32_to_f16<<<dim3((Cq * C + 255) / 256), 256, 0, stream>>>(Wq, WqH, (size_t)Cq * C);
  f32_to_f16<<<dim3((Cq * C + 255) / 256), 256, 0, stream>>>(Wk, WkH, (size_t)Cq * C);
  f32_to_f16<<<dim3((C * C + 255) / 256), 256, 0, stream>>>(Wv, WvH, (size_t)C * C);
  f32_to_f16<<<dim3((C * C + 255) / 256), 256, 0, stream>>>(Wout, WoH, (size_t)C * C);
  transpose_f32_to_f16<<<dim3(N / 32, C / 32, B), dim3(32, 8), 0, stream>>>(x, xhT, C, N);

  // --- q/k/v projections (WMMA) ---
  gemm_wmma_f16<true, true, true><<<dim3(N / 128, Cq / 32, B), 256, 0, stream>>>(
      WqH, 0, xhT, (size_t)N * C, bq, qT, (size_t)N * Cq, Cq, N, C);
  gemm_wmma_f16<true, true, true><<<dim3(N / 128, Cq / 32, B), 256, 0, stream>>>(
      WkH, 0, xhT, (size_t)N * C, bk, kTm, (size_t)N * Cq, Cq, N, C);
  gemm_wmma_f16<true, false, true><<<dim3(N / 128, C / 32, B), 256, 0, stream>>>(
      WvH, 0, xhT, (size_t)N * C, bv, vH, (size_t)C * N, C, N, C);

  // --- PAM flash attention (WMMA, streamed softmax) ---
  pam_flash<<<dim3(N / 16, B), 256, 0, stream>>>(qT, kTm, vH, pamH, C, N, Cq);

  // --- CAM: energy = x x^T, softmin softmax, apply (WMMA) ---
  gemm_wmma_f16<false, false, false><<<dim3(C / 128, C / 32, B), 256, 0, stream>>>(
      xh, (size_t)C * N, xh, (size_t)C * N, nullptr, ener, (size_t)C * C, C, C, N);
  cam_softmax<<<dim3(C, B), 256, 0, stream>>>(ener, attnH, C);
  gemm_wmma_f16<true, false, false><<<dim3(N / 128, C / 32, B), 256, 0, stream>>>(
      attnH, (size_t)C * C, xhT, (size_t)N * C, nullptr, camH, (size_t)C * N, C, N, C);

  // --- combine branches + residuals into conv operand ---
  combine_kernel<<<dim3(N / 32, C / 32, B), dim3(32, 8), 0, stream>>>(
      pamH, camH, x, gp, gc, featT, C, N);

  // --- 1x1 conv (WMMA) ---
  gemm_wmma_f16<false, false, false><<<dim3(N / 128, C / 32, B), 256, 0, stream>>>(
      WoH, 0, featT, (size_t)N * C, nullptr, convO, (size_t)C * N, C, N, C);

  // --- BatchNorm (training-mode batch stats) + ReLU ---
  bn_stats<<<dim3(C), 256, 0, stream>>>(convO, mean, rstd, B, C, N);
  size_t tot = (size_t)B * C * N;
  bn_apply<<<dim3((unsigned)((tot + 255) / 256)), 256, 0, stream>>>(
      convO, mean, rstd, bng, bnb, (float*)d_out, C, N, tot);
}
